// DecoderAttentionLSTM_42846593745223
// MI455X (gfx1250) — matrix operand order repaired and checked by reading, not compiled, not run on hardware
//
#include <hip/hip_runtime.h>
#include <hip/hip_bf16.h>

#define B_ 64
#define S_ 256
#define U_ 1024
#define T_ 512

typedef __bf16 bf16_t;
typedef __attribute__((ext_vector_type(16))) __bf16 v16bf;
typedef __attribute__((ext_vector_type(8)))  __bf16 v8bf;
typedef __attribute__((ext_vector_type(8)))  float  v8f;

__device__ __forceinline__ bf16_t f2bf(float f) {
  unsigned u = __builtin_bit_cast(unsigned, f);
  u += 0x7FFFu + ((u >> 16) & 1u);            // round-to-nearest-even
  unsigned short h = (unsigned short)(u >> 16);
  return __builtin_bit_cast(bf16_t, h);
}
__device__ __forceinline__ float bf2f(bf16_t b) {
  unsigned short s = __builtin_bit_cast(unsigned short, b);
  unsigned u = ((unsigned)s) << 16;
  return __builtin_bit_cast(float, u);
}
__device__ __forceinline__ float sigm_(float x) { return 1.0f / (1.0f + __expf(-x)); }

// ---------------------------------------------------------------------------
// Fragment helpers (CDNA5 16-bit WMMA layouts, wave32)
// A 16x32 bf16: lanes0-15 row M=lane, K {0..7,16..23}; lanes16-31 K {8..15,24..31}
// B 32x16 bf16 from WT (N-major, row n = original column): lane holds col (lane&15),
//   K {0..7,16..23} for lanes0-15; K {8..15,24..31} for lanes16-31.
// C 16x16 f32: VGPR v -> M = v + 8*(lane>>4), N = lane&15.
// ---------------------------------------------------------------------------

__device__ __forceinline__ v16bf load_a_f32(const float* __restrict__ A, int lda,
                                            int m0, int kb, int lane) {
  int row = m0 + (lane & 15);
  int k0  = (lane >> 4) * 8;
  const float* p = A + (size_t)row * lda + kb + k0;
  v16bf a;
#pragma unroll
  for (int j = 0; j < 8; ++j) a[j] = f2bf(p[j]);
#pragma unroll
  for (int j = 0; j < 8; ++j) a[8 + j] = f2bf(p[16 + j]);
  return a;
}

__device__ __forceinline__ v16bf load_a_bf16(const bf16_t* __restrict__ A, int lda,
                                             int m0, int kb, int lane) {
  int row = m0 + (lane & 15);
  int k0  = (lane >> 4) * 8;
  const bf16_t* p = A + (size_t)row * lda + kb + k0;
  v8bf lo = *(const v8bf*)(p);
  v8bf hi = *(const v8bf*)(p + 16);
  v16bf a;
#pragma unroll
  for (int j = 0; j < 8; ++j) { a[j] = lo[j]; a[8 + j] = hi[j]; }
  return a;
}

__device__ __forceinline__ v16bf load_b_wt(const bf16_t* __restrict__ WT, int K,
                                           int ncol, int kb, int lane) {
  int n  = ncol + (lane & 15);
  int kk = (lane >> 4) * 8;
  const bf16_t* p = WT + (size_t)n * K + kb + kk;
  v8bf lo = *(const v8bf*)(p);
  v8bf hi = *(const v8bf*)(p + 16);
  v16bf b;
#pragma unroll
  for (int j = 0; j < 8; ++j) { b[j] = lo[j]; b[8 + j] = hi[j]; }
  return b;
}

// ---------------------------------------------------------------------------
// One-time prep kernels
// ---------------------------------------------------------------------------
__global__ void k_f32_to_bf16(const float* __restrict__ src, bf16_t* __restrict__ dst, int n) {
  int i = blockIdx.x * blockDim.x + threadIdx.x;
  if (i < n) dst[i] = f2bf(src[i]);
}

// dst (C x R) = transpose of src (R x C), fp32 -> bf16
__global__ void k_transpose_bf16(const float* __restrict__ src, bf16_t* __restrict__ dst,
                                 int R, int C) {
  int i = blockIdx.x * blockDim.x + threadIdx.x;
  if (i >= R * C) return;
  int r = i / C, c = i % C;
  dst[(size_t)c * R + r] = f2bf(src[i]);
}

// h_proj(bf16) = h_bf16 (16384x1024) @ We1_h ; no bias/activation
__global__ void k_hproj(const bf16_t* __restrict__ hA, const bf16_t* __restrict__ We1hT,
                        bf16_t* __restrict__ hproj) {
  int waveId = blockIdx.x * (blockDim.x >> 5) + (threadIdx.x >> 5);
  int lane   = threadIdx.x & 31;
  int m0 = (waveId & 1023) * 16;        // 1024 M-tiles
  int n0 = (waveId >> 10) * 64;         // 16 N-tiles
  v8f c[4] = {};
  for (int kb = 0; kb < U_; kb += 32) {
    v16bf a = load_a_bf16(hA, U_, m0, kb, lane);
#pragma unroll
    for (int nf = 0; nf < 4; ++nf) {
      v16bf b = load_b_wt(We1hT, U_, n0 + nf * 16, kb, lane);
      c[nf] = __builtin_amdgcn_wmma_f32_16x16x32_bf16(false, a, false, b, (short)0, c[nf],
                                                      false, false);
    }
  }
  int crow  = m0 + ((lane >> 4) * 8);
  int ccol0 = n0 + (lane & 15);
#pragma unroll
  for (int nf = 0; nf < 4; ++nf) {
    int col = ccol0 + nf * 16;
#pragma unroll
    for (int v = 0; v < 8; ++v)
      hproj[(size_t)(crow + v) * U_ + col] = f2bf(c[nf][v]);
  }
}

// ---------------------------------------------------------------------------
// Per-step kernels
// ---------------------------------------------------------------------------

// u1ps (64x2048) = s (64x1024) @ [Wy1 | We1_s] ; cols<1024: tanh(+by1) = u1, else +be1 = ps
__global__ void k_gemm1(const float* __restrict__ s, const bf16_t* __restrict__ WcatT,
                        const float* __restrict__ by1, const float* __restrict__ be1,
                        float* __restrict__ out) {
  int waveId = blockIdx.x * (blockDim.x >> 5) + (threadIdx.x >> 5);
  int lane   = threadIdx.x & 31;
  int m0 = (waveId & 3) * 16;           // 4 M-tiles
  int n0 = (waveId >> 2) * 64;          // 32 N-tiles
  v8f c[4] = {};
  for (int kb = 0; kb < U_; kb += 32) {
    v16bf a = load_a_f32(s, U_, m0, kb, lane);
#pragma unroll
    for (int nf = 0; nf < 4; ++nf) {
      v16bf b = load_b_wt(WcatT, U_, n0 + nf * 16, kb, lane);
      c[nf] = __builtin_amdgcn_wmma_f32_16x16x32_bf16(false, a, false, b, (short)0, c[nf],
                                                      false, false);
    }
  }
  int crow  = m0 + ((lane >> 4) * 8);
  int ccol0 = n0 + (lane & 15);
#pragma unroll
  for (int nf = 0; nf < 4; ++nf) {
    int col = ccol0 + nf * 16;
    bool isY   = (col < U_);
    float bias = isY ? by1[col] : be1[col - U_];
#pragma unroll
    for (int v = 0; v < 8; ++v) {
      float val = c[nf][v] + bias;
      if (isY) val = tanhf(val);
      out[(size_t)(crow + v) * 2048 + col] = val;
    }
  }
}

// y (64x512) = u1 (cols 0..1023 of u1ps, stride 2048) @ Wy2 + by2 ; also writes d_out[:, t, :]
__global__ void k_gemm2(const float* __restrict__ u1ps, const bf16_t* __restrict__ Wy2T,
                        const float* __restrict__ by2, float* __restrict__ ybuf,
                        float* __restrict__ out_y, int t) {
  int waveId = blockIdx.x * (blockDim.x >> 5) + (threadIdx.x >> 5);
  int lane   = threadIdx.x & 31;
  int m0 = (waveId & 3) * 16;           // 4 M-tiles
  int n0 = (waveId >> 2) * 64;          // 8 N-tiles
  v8f c[4] = {};
  for (int kb = 0; kb < U_; kb += 32) {
    v16bf a = load_a_f32(u1ps, 2048, m0, kb, lane);
#pragma unroll
    for (int nf = 0; nf < 4; ++nf) {
      v16bf b = load_b_wt(Wy2T, U_, n0 + nf * 16, kb, lane);
      c[nf] = __builtin_amdgcn_wmma_f32_16x16x32_bf16(false, a, false, b, (short)0, c[nf],
                                                      false, false);
    }
  }
  int crow  = m0 + ((lane >> 4) * 8);
  int ccol0 = n0 + (lane & 15);
#pragma unroll
  for (int nf = 0; nf < 4; ++nf) {
    int col = ccol0 + nf * 16;
    float bias = by2[col];
#pragma unroll
    for (int v = 0; v < 8; ++v) {
      float val = c[nf][v] + bias;
      int row = crow + v;
      ybuf[(size_t)row * T_ + col] = val;
      out_y[(size_t)row * S_ * T_ + (size_t)t * T_ + col] = val;
    }
  }
}

// Z (64x4096) = sigmoid/tanh([y|s] @ [Wi|Wf|Wo|Wg] + b)
__global__ void k_gates(const float* __restrict__ ybuf, const float* __restrict__ s,
                        const bf16_t* __restrict__ WgT, const float* __restrict__ b_i,
                        const float* __restrict__ b_f, const float* __restrict__ b_o,
                        const float* __restrict__ b_g, float* __restrict__ Z) {
  int waveId = blockIdx.x * (blockDim.x >> 5) + (threadIdx.x >> 5);
  int lane   = threadIdx.x & 31;
  int m0 = (waveId & 3) * 16;           // 4 M-tiles
  int n0 = (waveId >> 2) * 64;          // 64 N-tiles
  const int K = T_ + U_;                // 1536
  v8f c[4] = {};
  for (int kb = 0; kb < K; kb += 32) {
    v16bf a = (kb < T_) ? load_a_f32(ybuf, T_, m0, kb, lane)
                        : load_a_f32(s, U_, m0, kb - T_, lane);
#pragma unroll
    for (int nf = 0; nf < 4; ++nf) {
      v16bf b = load_b_wt(WgT, K, n0 + nf * 16, kb, lane);
      c[nf] = __builtin_amdgcn_wmma_f32_16x16x32_bf16(false, a, false, b, (short)0, c[nf],
                                                      false, false);
    }
  }
  int crow  = m0 + ((lane >> 4) * 8);
  int ccol0 = n0 + (lane & 15);
#pragma unroll
  for (int nf = 0; nf < 4; ++nf) {
    int col  = ccol0 + nf * 16;
    int gate = col >> 10;               // 0=i 1=f 2=o 3=g
    int gc   = col & 1023;
    float bias = (gate == 0) ? b_i[gc] : (gate == 1) ? b_f[gc] : (gate == 2) ? b_o[gc] : b_g[gc];
#pragma unroll
    for (int v = 0; v < 8; ++v) {
      float val = c[nf][v] + bias;
      val = (gate == 3) ? tanhf(val) : sigm_(val);
      Z[(size_t)(crow + v) * 4096 + col] = val;
    }
  }
}

// Attention: one block per batch. e = sigmoid(sum_u sigmoid(hproj+ps)*We2 + be2),
// a = softmax_s(e), c = sum_s a*h. h/hproj in bf16 (L2-resident).
__global__ void k_attn(const bf16_t* __restrict__ h_bf, const bf16_t* __restrict__ hproj,
                       const float* __restrict__ u1ps, const float* __restrict__ We2,
                       const float* __restrict__ be2, float* __restrict__ c_out) {
  int b    = blockIdx.x;
  int tid  = threadIdx.x;               // 0..255
  int lane = tid & 31;
  int wave = tid >> 5;                  // 0..7
  __shared__ float sh_e[S_];
  __shared__ float sh_red[8];

  const bf16_t* hp_b = hproj + (size_t)b * S_ * U_;
  const float*  ps_b = u1ps + (size_t)b * 2048 + U_;   // ps part of u1ps
  float be2v = be2[0];

  for (int sp = wave; sp < S_; sp += 8) {
    const bf16_t* row = hp_b + (size_t)sp * U_;
    float acc = 0.f;
    for (int u = lane; u < U_; u += 32)
      acc += sigm_(bf2f(row[u]) + ps_b[u]) * We2[u];
#pragma unroll
    for (int off = 16; off > 0; off >>= 1) acc += __shfl_xor(acc, off, 32);
    if (lane == 0) sh_e[sp] = acc + be2v;
  }
  __syncthreads();

  float e = sigm_(sh_e[tid]);           // logit of softmax
  float m = e;
#pragma unroll
  for (int off = 16; off > 0; off >>= 1) m = fmaxf(m, __shfl_xor(m, off, 32));
  if (lane == 0) sh_red[wave] = m;
  __syncthreads();
  float mall = sh_red[0];
#pragma unroll
  for (int i = 1; i < 8; ++i) mall = fmaxf(mall, sh_red[i]);
  float ex = __expf(e - mall);
  float ss = ex;
#pragma unroll
  for (int off = 16; off > 0; off >>= 1) ss += __shfl_xor(ss, off, 32);
  __syncthreads();
  if (lane == 0) sh_red[wave] = ss;
  __syncthreads();
  float tot = 0.f;
#pragma unroll
  for (int i = 0; i < 8; ++i) tot += sh_red[i];
  sh_e[tid] = ex / tot;                 // softmax weight a[s]
  __syncthreads();

  const bf16_t* h_b = h_bf + (size_t)b * S_ * U_;
  for (int u = tid; u < U_; u += 256) {
    float acc = 0.f;
    for (int sp = 0; sp < S_; ++sp)
      acc += sh_e[sp] * bf2f(h_b[(size_t)sp * U_ + u]);
    c_out[(size_t)b * U_ + u] = acc;
  }
}

// s_next = o * tanh(f*c_att + i*g)
__global__ void k_point(const float* __restrict__ Z, const float* __restrict__ c_att,
                        float* __restrict__ s_next) {
  int idx = blockIdx.x * blockDim.x + threadIdx.x;   // < 64*1024
  int b = idx >> 10, u = idx & 1023;
  const float* z = Z + (size_t)b * 4096;
  float i = z[u], f = z[1024 + u], o = z[2048 + u], g = z[3072 + u];
  float c = f * c_att[idx] + i * g;
  s_next[idx] = o * tanhf(c);
}

// ---------------------------------------------------------------------------
extern "C" void kernel_launch(void* const* d_in, const int* in_sizes, int n_in,
                              void* d_out, int out_size, void* d_ws, size_t ws_size,
                              hipStream_t stream) {
  const float* h   = (const float*)d_in[0];
  const float* s0  = (const float*)d_in[1];
  const float* Wy1 = (const float*)d_in[2];
  const float* by1 = (const float*)d_in[3];
  const float* Wy2 = (const float*)d_in[4];
  const float* by2 = (const float*)d_in[5];
  const float* We1 = (const float*)d_in[6];
  const float* be1 = (const float*)d_in[7];
  const float* We2 = (const float*)d_in[8];
  const float* be2 = (const float*)d_in[9];
  const float* Wfm = (const float*)d_in[10];
  const float* bfv = (const float*)d_in[11];
  const float* Wim = (const float*)d_in[12];
  const float* biv = (const float*)d_in[13];
  const float* Wgm = (const float*)d_in[14];
  const float* bgv = (const float*)d_in[15];
  const float* Wom = (const float*)d_in[16];
  const float* bov = (const float*)d_in[17];
  float* out = (float*)d_out;
  (void)in_sizes; (void)n_in; (void)out_size; (void)ws_size;

  char* ws = (char*)d_ws;
  size_t off = 0;
  auto alloc = [&](size_t bytes) -> void* {
    void* p = ws + off;
    off = (off + bytes + 255) & ~(size_t)255;
    return p;
  };
  const int KG = T_ + U_;                                   // 1536
  bf16_t* h_bf   = (bf16_t*)alloc((size_t)B_ * S_ * U_ * 2);
  bf16_t* hproj  = (bf16_t*)alloc((size_t)B_ * S_ * U_ * 2);
  bf16_t* We1hT  = (bf16_t*)alloc((size_t)U_ * U_ * 2);
  bf16_t* WcatT  = (bf16_t*)alloc((size_t)2 * U_ * U_ * 2); // [Wy1^T ; We1_s^T]
  bf16_t* Wy2T   = (bf16_t*)alloc((size_t)T_ * U_ * 2);
  bf16_t* WgT    = (bf16_t*)alloc((size_t)4 * U_ * KG * 2); // [Wi|Wf|Wo|Wg]^T
  float*  sA     = (float*)alloc((size_t)B_ * U_ * 4);
  float*  sB     = (float*)alloc((size_t)B_ * U_ * 4);
  float*  u1ps   = (float*)alloc((size_t)B_ * 2048 * 4);
  float*  ybuf   = (float*)alloc((size_t)B_ * T_ * 4);
  float*  Z      = (float*)alloc((size_t)B_ * 4096 * 4);
  float*  c_att  = (float*)alloc((size_t)B_ * U_ * 4);

  // --- one-time prep ---
  {
    int nh = B_ * S_ * U_;
    k_f32_to_bf16<<<(nh + 255) / 256, 256, 0, stream>>>(h, h_bf, nh);
    int nUU = U_ * U_;
    k_transpose_bf16<<<(nUU + 255) / 256, 256, 0, stream>>>(Wy1, WcatT, U_, U_);
    k_transpose_bf16<<<(nUU + 255) / 256, 256, 0, stream>>>(We1 + (size_t)U_ * U_,
                                                            WcatT + (size_t)U_ * U_, U_, U_);
    k_transpose_bf16<<<(nUU + 255) / 256, 256, 0, stream>>>(We1, We1hT, U_, U_);
    int nUT = U_ * T_;
    k_transpose_bf16<<<(nUT + 255) / 256, 256, 0, stream>>>(Wy2, Wy2T, U_, T_);
    int nG = KG * U_;
    k_transpose_bf16<<<(nG + 255) / 256, 256, 0, stream>>>(Wim, WgT, KG, U_);
    k_transpose_bf16<<<(nG + 255) / 256, 256, 0, stream>>>(Wfm, WgT + (size_t)1 * U_ * KG, KG, U_);
    k_transpose_bf16<<<(nG + 255) / 256, 256, 0, stream>>>(Wom, WgT + (size_t)2 * U_ * KG, KG, U_);
    k_transpose_bf16<<<(nG + 255) / 256, 256, 0, stream>>>(Wgm, WgT + (size_t)3 * U_ * KG, KG, U_);
    k_hproj<<<4096, 128, 0, stream>>>(h_bf, We1hT, hproj);   // 16384 waves
    hipMemcpyAsync(sA, s0, (size_t)B_ * U_ * 4, hipMemcpyDeviceToDevice, stream);
  }

  // --- sequential scan over S steps ---
  float* scur = sA;
  float* snxt = sB;
  for (int t = 0; t < S_; ++t) {
    k_gemm1<<<32, 128, 0, stream>>>(scur, WcatT, by1, be1, u1ps);
    k_gemm2<<<8, 128, 0, stream>>>(u1ps, Wy2T, by2, ybuf, out, t);
    k_attn<<<B_, 256, 0, stream>>>(h_bf, hproj, u1ps, We2, be2, c_att);
    k_gates<<<64, 128, 0, stream>>>(ybuf, scur, WgT, biv, bfv, bov, bgv, Z);
    k_point<<<(B_ * U_) / 256, 256, 0, stream>>>(Z, c_att, snxt);
    float* tmp = scur; scur = snxt; snxt = tmp;
  }
}
